// Block_ssmamba_4020089389066
// MI455X (gfx1250) — matrix operand order repaired and checked by reading, not compiled, not run on hardware
//
#include <hip/hip_runtime.h>
#include <hip/hip_bf16.h>

typedef __attribute__((ext_vector_type(16))) _Float16 v16h;
typedef __attribute__((ext_vector_type(8)))  float    v8f;

#define BB    2
#define CC    128
#define HH    64
#define WWI   64
#define HWSZ  4096      // H*W
#define TT    8192      // B*H*W tokens
#define NST   16        // SSM state size N
#define KDIR  2
#define RSPA  8
#define CNE   16        // CN
#define GC    8
#define XLD   48        // padded x_dbl width (8 + 16 + 16 = 40 -> 48)

// ---------------------------------------------------------------------------
// WMMA GEMM with N-strip reuse: each wave computes a 16 x (NT*16) strip.
// C[M,N] = A[M,K](f16 row-major) x B[N,K](f16 row-major)^T
// flags bit0: accumulate into Cout; bit1: NCHW scatter (token m -> b,hw).
// addend (optional): row-major [M,N] f32 added to the result.
// ---------------------------------------------------------------------------
template <int NT>
__global__ __launch_bounds__(128) void k_wmma_gemm(
    const _Float16* __restrict__ A, const _Float16* __restrict__ Bw,
    float* __restrict__ Cout, const float* __restrict__ bias,
    const float* __restrict__ addend,
    int M, int N, int Kd, int ldc, int flags, int HW)
{
  int tilesM = M >> 4;
  int groupsN = (N >> 4) / NT;
  int wid = blockIdx.x * 4 + (threadIdx.x >> 5);
  if (wid >= tilesM * groupsN) return;          // whole-wave uniform exit
  int tm = wid % tilesM, tg = wid / tilesM;
  int lane = threadIdx.x & 31;
  int hi = lane >> 4;
  int row = lane & 15;

  const _Float16* arow = A + (size_t)(tm * 16 + row) * Kd;
  const _Float16* brow[NT];
#pragma unroll
  for (int j = 0; j < NT; ++j)
    brow[j] = Bw + (size_t)((tg * NT + j) * 16 + row) * Kd;

  v8f acc[NT];
#pragma unroll
  for (int j = 0; j < NT; ++j) acc[j] = (v8f){};

  for (int kk = 0; kk < Kd; kk += 32) {
    v16h a;
    // A 16x32 f16 layout: lane-half hi holds K = kk+16*hi..+15
#pragma unroll
    for (int i = 0; i < 8; ++i) {
      a[i]     = arow[kk + 16 * hi + i];
      a[8 + i] = arow[kk + 16 * hi + 8 + i];
    }
#pragma unroll
    for (int j = 0; j < NT; ++j) {
      v16h b;
      // B 32x16 f16 layout: lane = N col; lane-half hi holds K = kk+16*hi..+15
#pragma unroll
      for (int i = 0; i < 16; ++i) b[i] = brow[j][kk + 16 * hi + i];
      acc[j] = __builtin_amdgcn_wmma_f32_16x16x32_f16(false, a, false, b,
                                                      (short)0, acc[j], false, false);
    }
  }

#pragma unroll
  for (int j = 0; j < NT; ++j) {
    int n = (tg * NT + j) * 16 + (lane & 15);
    float bv = bias ? bias[n] : 0.f;
#pragma unroll
    for (int i = 0; i < 8; ++i) {
      int m = tm * 16 + i + (hi ? 8 : 0);
      float v = acc[j][i] + bv;
      if (addend) v += addend[(size_t)m * N + n];
      if (flags & 2) {
        int b_ = m / HW, hw = m % HW;
        size_t idx = ((size_t)b_ * N + n) * HW + hw;
        if (flags & 1) Cout[idx] += v; else Cout[idx] = v;
      } else {
        size_t idx = (size_t)m * ldc + n;
        if (flags & 1) Cout[idx] += v; else Cout[idx] = v;
      }
    }
  }
}

// ---------------------------------------------------------------------------
// x (B,C,H,W) f32 -> token-major (T,C) f16
// ---------------------------------------------------------------------------
__global__ void k_nchw_to_nhwc_h(const float* __restrict__ x, _Float16* __restrict__ xh)
{
  int idx = blockIdx.x * blockDim.x + threadIdx.x;
  if (idx >= BB * CC * HWSZ) return;
  int hw = idx & (HWSZ - 1);
  int c  = (idx >> 12) & (CC - 1);
  int b  = idx >> 19;
  xh[((size_t)b * HWSZ + hw) * CC + c] = (_Float16)x[idx];
}

__global__ void k_cvt_h(const float* __restrict__ in, _Float16* __restrict__ out, int n)
{
  int i = blockIdx.x * blockDim.x + threadIdx.x;
  if (i < n) out[i] = (_Float16)in[i];
}

// pack spa_xproj (2,40,128) -> (2,48,128) f16, zero rows 40..47
__global__ void k_pack_xproj(const float* __restrict__ xp, _Float16* __restrict__ out)
{
  int idx = blockIdx.x * blockDim.x + threadIdx.x;
  if (idx >= 2 * XLD * CC) return;
  int d = idx & (CC - 1);
  int c = (idx >> 7) % XLD;
  int k = idx / (XLD * CC);
  out[idx] = (c < 40) ? (_Float16)xp[((size_t)k * 40 + c) * CC + d] : (_Float16)0.f;
}

// ---------------------------------------------------------------------------
// depthwise 3x3 (SAME, zero-pad) + bias + SiLU on token-major (T,C) f32
// optionally emit f16 copy and per-batch L-flipped f16 copy (for scan dir k=1)
// ---------------------------------------------------------------------------
__global__ void k_dwconv_silu(const float* __restrict__ in, const float* __restrict__ w,
                              const float* __restrict__ bias, float* __restrict__ outf,
                              _Float16* __restrict__ outh, _Float16* __restrict__ outhf)
{
  int idx = blockIdx.x * blockDim.x + threadIdx.x;
  if (idx >= TT * CC) return;
  int c = idx & (CC - 1);
  int t = idx >> 7;
  int b = t >> 12;
  int l = t & (HWSZ - 1);
  int h = l >> 6, x = l & (WWI - 1);
  float acc = bias[c];
#pragma unroll
  for (int dy = -1; dy <= 1; ++dy) {
    int hy = h + dy;
    if (hy < 0 || hy >= HH) continue;
#pragma unroll
    for (int dx = -1; dx <= 1; ++dx) {
      int wx = x + dx;
      if (wx < 0 || wx >= WWI) continue;
      acc += w[c * 9 + (dy + 1) * 3 + (dx + 1)] *
             in[(((size_t)b * HWSZ) + hy * WWI + wx) * CC + c];
    }
  }
  float s = acc / (1.f + __expf(-acc));   // SiLU
  outf[idx] = s;
  if (outh)  outh[idx] = (_Float16)s;
  if (outhf) outhf[(((size_t)b * HWSZ) + (HWSZ - 1 - l)) * CC + c] = (_Float16)s;
}

// ---------------------------------------------------------------------------
// delta = softplus(dt_w @ dts + dt_b)   layout (b,k,d,L)
// ---------------------------------------------------------------------------
__global__ void k_delta_spa(const float* __restrict__ xdbl, const float* __restrict__ dtw,
                            const float* __restrict__ dtb, float* __restrict__ delta)
{
  int idx = blockIdx.x * blockDim.x + threadIdx.x;
  if (idx >= BB * KDIR * CC * HWSZ) return;
  int l  = idx & (HWSZ - 1);
  int d  = (idx >> 12) & (CC - 1);
  int bk = idx >> 19;
  int k  = bk & 1;
  const float* xr = xdbl + ((size_t)bk * HWSZ + l) * XLD;
  const float* wr = dtw + ((size_t)k * CC + d) * RSPA;
  float acc = dtb[k * CC + d];
#pragma unroll
  for (int r = 0; r < RSPA; ++r) acc += wr[r] * xr[r];
  delta[idx] = (acc > 20.f) ? acc : __logf(1.f + __expf(acc));
}

// ---------------------------------------------------------------------------
// Spatial selective scan. 256 waves; each wave = 2 channels of one (b,k):
// lane-half = channel, lane&15 = state n. Sequential over L=4096.
// ---------------------------------------------------------------------------
__global__ __launch_bounds__(32) void k_scan_spa(
    const float* __restrict__ delta,   // (b,k,d,L)
    const float* __restrict__ xdbl,    // (b,k,L,48)
    const float* __restrict__ u_nhwc,  // (b,L,C) f32 (silu'd x1)
    const float* __restrict__ Alog,    // (K,C,N)
    const float* __restrict__ Dvec,    // (K,C)
    float* __restrict__ ymid)          // (b,k,L,C)
{
  int w = blockIdx.x;                  // 0..255
  int lane = threadIdx.x & 31;
  int b = w >> 7, k = (w >> 6) & 1, dp = w & 63;
  int d = dp * 2 + (lane >> 4);
  int n = lane & 15;

  float Aneg = -__expf(Alog[((size_t)k * CC + d) * NST + n]);
  float Dd = Dvec[k * CC + d];
  const float* dptr = delta + (((size_t)(b * KDIR + k)) * CC + d) * HWSZ;
  const float* xdb  = xdbl  + ((size_t)(b * KDIR + k)) * HWSZ * XLD;
  float* yp = ymid + ((size_t)(b * KDIR + k)) * HWSZ * CC;

  float h = 0.f;
  for (int t = 0; t < HWSZ; ++t) {
    __builtin_prefetch(xdb + (size_t)(t + 8) * XLD, 0, 1);
    float dl = dptr[t];
    int lu = k ? (HWSZ - 1 - t) : t;
    float u = u_nhwc[((size_t)b * HWSZ + lu) * CC + d];
    float bn = xdb[(size_t)t * XLD + 8 + n];
    float cn = xdb[(size_t)t * XLD + 24 + n];
    h = __expf(dl * Aneg) * h + (dl * u) * bn;
    float pv = h * cn;
    pv += __shfl_xor(pv, 1, 16);
    pv += __shfl_xor(pv, 2, 16);
    pv += __shfl_xor(pv, 4, 16);
    pv += __shfl_xor(pv, 8, 16);
    if (n == 0) yp[(size_t)t * CC + d] = pv + Dd * u;
  }
}

// ---------------------------------------------------------------------------
// merge forward/backward scan + LayerNorm over C, emit f16 tokens for out-proj
// ---------------------------------------------------------------------------
__global__ __launch_bounds__(128) void k_merge_ln_spa(
    const float* __restrict__ ymid, const float* __restrict__ ng,
    const float* __restrict__ nb, _Float16* __restrict__ outh)
{
  int t = blockIdx.x;
  int b = t / HWSZ, l = t % HWSZ;
  int c = threadIdx.x;
  float v = ymid[(((size_t)(b * KDIR + 0)) * HWSZ + l) * CC + c]
          + ymid[(((size_t)(b * KDIR + 1)) * HWSZ + (HWSZ - 1 - l)) * CC + c];
  __shared__ float red[128];
  __shared__ float s_mu, s_rv;
  red[c] = v; __syncthreads();
  for (int s = 64; s > 0; s >>= 1) { if (c < s) red[c] += red[c + s]; __syncthreads(); }
  if (c == 0) s_mu = red[0] * (1.f / CC);
  __syncthreads();
  float mu = s_mu, dv = v - mu;
  red[c] = dv * dv; __syncthreads();
  for (int s = 64; s > 0; s >>= 1) { if (c < s) red[c] += red[c + s]; __syncthreads(); }
  if (c == 0) s_rv = rsqrtf(red[0] * (1.f / CC) + 1e-5f);
  __syncthreads();
  outh[(size_t)t * CC + c] = (_Float16)(dv * s_rv * ng[c] + nb[c]);
}

// ---------------------------------------------------------------------------
// Spectral branch: fully fused SS2D per token (scan length GC=8, parallel).
// One wave per token; lane = (k, d) sequence; 16 states in registers.
// ---------------------------------------------------------------------------
__global__ __launch_bounds__(256) void k_spe_ss2d(
    const float* __restrict__ x1e,    // (T,C) f32 post dwconv+silu
    const float* __restrict__ xproj,  // (2,33,16)
    const float* __restrict__ dtw,    // (2,16,1)
    const float* __restrict__ dtb,    // (2,16)
    const float* __restrict__ Alog,   // (2,16,16)
    const float* __restrict__ Dvec,   // (2,16)
    const float* __restrict__ ng, const float* __restrict__ nb,  // (16)
    _Float16* __restrict__ outh)      // (T,C)
{
  __shared__ float s_xr[8][CC];
  __shared__ float s_dbl[8][2][33 * GC];
  __shared__ float s_y[8][CC];
  int tk = threadIdx.x >> 5;
  int lane = threadIdx.x & 31;
  int t = blockIdx.x * 8 + tk;

#pragma unroll
  for (int j = 0; j < 4; ++j)
    s_xr[tk][lane + 32 * j] = x1e[(size_t)t * CC + lane + 32 * j];
  __syncthreads();

  // x_dbl[k][c][g] = sum_d xproj[k,c,d] * xr[d, gg]  (gg flipped for k=1)
  for (int idx = lane; idx < 2 * 33 * GC; idx += 32) {
    int k = idx / (33 * GC);
    int rem = idx - k * (33 * GC);
    int c = rem >> 3, g = rem & 7;
    int gg = k ? (GC - 1 - g) : g;
    float acc = 0.f;
#pragma unroll
    for (int dd = 0; dd < CNE; ++dd)
      acc += xproj[((size_t)k * 33 + c) * CNE + dd] * s_xr[tk][dd * GC + gg];
    s_dbl[tk][k][c * GC + g] = acc;
  }
  __syncthreads();

  int k = lane >> 4, d = lane & 15;
  float Aneg[NST];
#pragma unroll
  for (int n = 0; n < NST; ++n)
    Aneg[n] = -__expf(Alog[((size_t)k * CNE + d) * NST + n]);
  float dtwv = dtw[k * CNE + d];
  float dtbv = dtb[k * CNE + d];
  float Dd   = Dvec[k * CNE + d];
  float hreg[NST];
#pragma unroll
  for (int n = 0; n < NST; ++n) hreg[n] = 0.f;

  float yloc[GC];
  for (int g = 0; g < GC; ++g) {
    float a = dtwv * s_dbl[tk][k][g] + dtbv;
    float dl = (a > 20.f) ? a : __logf(1.f + __expf(a));
    int gg = k ? (GC - 1 - g) : g;
    float u = s_xr[tk][d * GC + gg];
    float yv = 0.f;
#pragma unroll
    for (int n = 0; n < NST; ++n) {
      float bn = s_dbl[tk][k][(1 + n) * GC + g];
      float cn = s_dbl[tk][k][(1 + NST + n) * GC + g];
      hreg[n] = __expf(dl * Aneg[n]) * hreg[n] + (dl * u) * bn;
      yv += hreg[n] * cn;
    }
    yv += Dd * u;
    int pos = k ? (GC - 1 - g) : g;      // flip(y1) before merging
    yloc[pos] = yv;
  }
  if (k == 0) {
#pragma unroll
    for (int p = 0; p < GC; ++p) s_y[tk][d * GC + p] = yloc[p];
  }
  __syncthreads();
  if (k == 1) {
#pragma unroll
    for (int p = 0; p < GC; ++p) s_y[tk][d * GC + p] += yloc[p];
  }
  __syncthreads();

  // LayerNorm over d (16) per position g; channel index = d*GC + g
  if (lane < GC) {
    int g = lane;
    float mu = 0.f;
    for (int d2 = 0; d2 < CNE; ++d2) mu += s_y[tk][d2 * GC + g];
    mu *= (1.f / CNE);
    float var = 0.f;
    for (int d2 = 0; d2 < CNE; ++d2) {
      float dv = s_y[tk][d2 * GC + g] - mu;
      var += dv * dv;
    }
    float rv = rsqrtf(var * (1.f / CNE) + 1e-5f);
    for (int d2 = 0; d2 < CNE; ++d2)
      outh[(size_t)t * CC + d2 * GC + g] =
          (_Float16)((s_y[tk][d2 * GC + g] - mu) * rv * ng[d2] + nb[d2]);
  }
}

// ---------------------------------------------------------------------------
extern "C" void kernel_launch(void* const* d_in, const int* in_sizes, int n_in,
                              void* d_out, int out_size, void* d_ws, size_t ws_size,
                              hipStream_t stream)
{
  const float* x         = (const float*)d_in[0];
  const float* c1_w      = (const float*)d_in[7];
  const float* c1_b      = (const float*)d_in[8];
  const float* spa_in_w  = (const float*)d_in[9];
  const float* spa_dwc_w = (const float*)d_in[12];
  const float* spa_dwc_b = (const float*)d_in[13];
  const float* spa_out_w = (const float*)d_in[14];
  const float* spa_xproj = (const float*)d_in[15];
  const float* spa_dt_w  = (const float*)d_in[16];
  const float* spa_dt_b  = (const float*)d_in[17];
  const float* spa_Alog  = (const float*)d_in[18];
  const float* spa_D     = (const float*)d_in[19];
  const float* spa_ng    = (const float*)d_in[20];
  const float* spa_nb    = (const float*)d_in[21];
  const float* spe_in_w  = (const float*)d_in[22];
  const float* spe_dwc_w = (const float*)d_in[25];
  const float* spe_dwc_b = (const float*)d_in[26];
  const float* spe_out_w = (const float*)d_in[27];
  const float* spe_xproj = (const float*)d_in[28];
  const float* spe_dt_w  = (const float*)d_in[29];
  const float* spe_dt_b  = (const float*)d_in[30];
  const float* spe_Alog  = (const float*)d_in[31];
  const float* spe_D     = (const float*)d_in[32];
  const float* spe_ng    = (const float*)d_in[33];
  const float* spe_nb    = (const float*)d_in[34];
  float* out = (float*)d_out;

  char* wsp = (char*)d_ws;
  size_t off = 0;
  auto carve = [&](size_t bytes) -> void* {
    off = (off + 255) & ~(size_t)255;
    void* p = wsp + off;
    off += bytes;
    return p;
  };

  _Float16* xh        = (_Float16*)carve((size_t)TT * CC * 2);
  _Float16* w_in_a_h  = (_Float16*)carve((size_t)CC * CC * 2);
  _Float16* w_in_e_h  = (_Float16*)carve((size_t)CC * CC * 2);
  _Float16* w_out_a_h = (_Float16*)carve((size_t)CC * CC * 2);
  _Float16* w_out_e_h = (_Float16*)carve((size_t)CC * CC * 2);
  _Float16* w_c1_h    = (_Float16*)carve((size_t)CC * CC * 2);
  _Float16* xprojpad  = (_Float16*)carve((size_t)2 * XLD * CC * 2);
  float*    x1a_pre   = (float*)carve((size_t)TT * CC * 4);
  float*    x1e_pre   = (float*)carve((size_t)TT * CC * 4);
  float*    x1a_f     = (float*)carve((size_t)TT * CC * 4);
  _Float16* x1a_h     = (_Float16*)carve((size_t)TT * CC * 2);
  _Float16* x1a_hf    = (_Float16*)carve((size_t)TT * CC * 2);
  float*    x1e_f     = (float*)carve((size_t)TT * CC * 4);
  float*    xdbl      = (float*)carve((size_t)BB * KDIR * HWSZ * XLD * 4);
  float*    delta     = (float*)carve((size_t)BB * KDIR * CC * HWSZ * 4);
  float*    ymid      = (float*)carve((size_t)BB * KDIR * HWSZ * CC * 4);
  _Float16* yah       = (_Float16*)carve((size_t)TT * CC * 2);
  _Float16* yeh       = (_Float16*)carve((size_t)TT * CC * 2);
  float*    sbuf      = (float*)carve((size_t)TT * CC * 4);
  _Float16* sh        = (_Float16*)carve((size_t)TT * CC * 2);
  (void)ws_size; (void)n_in; (void)in_sizes; (void)out_size;

  const int TC = TT * CC;

  // ---- format prep
  k_nchw_to_nhwc_h<<<(TC + 255) / 256, 256, 0, stream>>>(x, xh);
  k_cvt_h<<<(CC * CC + 255) / 256, 256, 0, stream>>>(spa_in_w, w_in_a_h, CC * CC);
  k_cvt_h<<<(CC * CC + 255) / 256, 256, 0, stream>>>(spe_in_w, w_in_e_h, CC * CC);
  k_cvt_h<<<(CC * CC + 255) / 256, 256, 0, stream>>>(spa_out_w, w_out_a_h, CC * CC);
  k_cvt_h<<<(CC * CC + 255) / 256, 256, 0, stream>>>(spe_out_w, w_out_e_h, CC * CC);
  k_cvt_h<<<(CC * CC + 255) / 256, 256, 0, stream>>>(c1_w, w_c1_h, CC * CC);
  k_pack_xproj<<<(2 * XLD * CC + 255) / 256, 256, 0, stream>>>(spa_xproj, xprojpad);

  // ---- input projections (WMMA, 16x64 strip per wave)
  {
    int waves = (TT / 16) * ((CC / 16) / 4);      // 512 * 2 = 1024
    int grid = (waves + 3) / 4;
    k_wmma_gemm<4><<<grid, 128, 0, stream>>>(xh, w_in_a_h, x1a_pre, nullptr, nullptr,
                                             TT, CC, CC, CC, 0, HWSZ);
    k_wmma_gemm<4><<<grid, 128, 0, stream>>>(xh, w_in_e_h, x1e_pre, nullptr, nullptr,
                                             TT, CC, CC, CC, 0, HWSZ);
  }

  // ---- depthwise 3x3 + SiLU
  k_dwconv_silu<<<(TC + 255) / 256, 256, 0, stream>>>(x1a_pre, spa_dwc_w, spa_dwc_b,
                                                      x1a_f, x1a_h, x1a_hf);
  k_dwconv_silu<<<(TC + 255) / 256, 256, 0, stream>>>(x1e_pre, spe_dwc_w, spe_dwc_b,
                                                      x1e_f, nullptr, nullptr);

  // ---- x_dbl GEMMs (per batch, per direction): 16x48 strip per wave
  {
    int waves = (HWSZ / 16) * ((XLD / 16) / 3);   // 256
    int grid = (waves + 3) / 4;
    for (int b = 0; b < BB; ++b)
      for (int k = 0; k < KDIR; ++k) {
        const _Float16* A = (k == 0 ? x1a_h : x1a_hf) + (size_t)b * HWSZ * CC;
        const _Float16* Bw = xprojpad + (size_t)k * XLD * CC;
        float* Cp = xdbl + ((size_t)(b * KDIR + k)) * HWSZ * XLD;
        k_wmma_gemm<3><<<grid, 128, 0, stream>>>(A, Bw, Cp, nullptr, nullptr,
                                                 HWSZ, XLD, CC, XLD, 0, HWSZ);
      }
  }

  // ---- delta = softplus(dt_w @ dts + dt_b)
  k_delta_spa<<<(BB * KDIR * CC * HWSZ + 255) / 256, 256, 0, stream>>>(
      xdbl, spa_dt_w, spa_dt_b, delta);

  // ---- spatial selective scan (latency-bound core)
  k_scan_spa<<<256, 32, 0, stream>>>(delta, xdbl, x1a_f, spa_Alog, spa_D, ymid);

  // ---- merge dirs + LayerNorm
  k_merge_ln_spa<<<TT, 128, 0, stream>>>(ymid, spa_ng, spa_nb, yah);

  // ---- spectral branch fused SS2D
  k_spe_ss2d<<<TT / 8, 256, 0, stream>>>(x1e_f, spe_xproj, spe_dt_w, spe_dt_b,
                                         spe_Alog, spe_D, spe_ng, spe_nb, yeh);

  // ---- output projections: s = yah @ spa_out^T + yeh @ spe_out^T (WMMA)
  {
    int waves = (TT / 16) * ((CC / 16) / 4);
    int grid = (waves + 3) / 4;
    k_wmma_gemm<4><<<grid, 128, 0, stream>>>(yah, w_out_a_h, sbuf, nullptr, nullptr,
                                             TT, CC, CC, CC, 0, HWSZ);
    k_wmma_gemm<4><<<grid, 128, 0, stream>>>(yeh, w_out_e_h, sbuf, nullptr, nullptr,
                                             TT, CC, CC, CC, 1 /*accumulate*/, HWSZ);
  }

  // ---- final: out = s + conv1x1(s, c1_w) + c1_b, scattered to NCHW
  k_cvt_h<<<(TC + 255) / 256, 256, 0, stream>>>(sbuf, sh, TC);
  {
    int waves = (TT / 16) * ((CC / 16) / 4);
    int grid = (waves + 3) / 4;
    k_wmma_gemm<4><<<grid, 128, 0, stream>>>(sh, w_c1_h, out, c1_b, sbuf,
                                             TT, CC, CC, CC, 2 /*NCHW*/, HWSZ);
  }
}
